// PQCodebookModel_59854664237421
// MI455X (gfx1250) — matrix operand order repaired and checked by reading, not compiled, not run on hardware
//
#include <hip/hip_runtime.h>

typedef __attribute__((ext_vector_type(16))) __bf16 v16bf;
typedef __attribute__((ext_vector_type(8)))  __bf16 v8bf;
typedef __attribute__((ext_vector_type(8)))  float  v8f;

#define N_ROWS 65536   // 16 * 4096
#define EDIM   768     // 8 * 96
#define SUBD   96
#define NCODE  256
#define NSUB   8

__global__ __launch_bounds__(128, 1) void pq_wmma_kernel(
    const float* __restrict__ emb, const float* __restrict__ cb,
    float* __restrict__ out)
{
    // LDS: codebook for this subspace split into bf16 hi/lo + squared norms.
    // 2*49152 + 1024 = 99328 bytes (fits 320KB WGP LDS easily).
    __shared__ __align__(16) __bf16 cb_hi[NCODE * SUBD];
    __shared__ __align__(16) __bf16 cb_lo[NCODE * SUBD];
    __shared__ __align__(16) float  c2s[NCODE];

    const int m   = blockIdx.y;
    const int tid = threadIdx.x;

    // ---- Stage codebook into LDS (hi/lo bf16) and compute ||c||^2 ----
    for (int k = tid; k < NCODE; k += 128) {
        const float* src = cb + ((size_t)m * NCODE + k) * SUBD;
        float s2 = 0.f;
        for (int j = 0; j < SUBD; j += 4) {
            float4 v = *(const float4*)(src + j);
            float vv[4] = {v.x, v.y, v.z, v.w};
#pragma unroll
            for (int q = 0; q < 4; ++q) {
                float f = vv[q];
                s2 += f * f;
                __bf16 h = (__bf16)f;
                cb_hi[k * SUBD + j + q] = h;
                cb_lo[k * SUBD + j + q] = (__bf16)(f - (float)h);
            }
        }
        c2s[k] = s2;
    }
    __syncthreads();

    const int lane = tid & 31;
    const int wave = tid >> 5;
    const int hl   = lane >> 4;   // lane half (wave32)
    const int lm   = lane & 15;

    // Each wave handles a pair of 16-row tiles per iteration (32 rows),
    // so every A-fragment LDS load feeds 6 WMMAs.
    for (int p = blockIdx.x * 4 + wave; p < (N_ROWS / 32); p += gridDim.x * 4) {
        // ---- Build B fragments (embeddings) in registers, bf16 hi/lo split ----
        // B layout (16-bit 32x16): lane holds column N=lane&15, K = 16*hl + e.
        v16bf Bhi[2][3], Blo[2][3];
        int rowbase[2];
#pragma unroll
        for (int t = 0; t < 2; ++t) {
            rowbase[t] = (p * 2 + t) * 16;
            const float* xrow = emb + (size_t)(rowbase[t] + lm) * EDIM + m * SUBD;
#pragma unroll
            for (int c = 0; c < 3; ++c) {
                const float* ptr = xrow + c * 32 + hl * 16;
#pragma unroll
                for (int q = 0; q < 4; ++q) {
                    float4 v = *(const float4*)(ptr + q * 4);
                    float vv[4] = {v.x, v.y, v.z, v.w};
#pragma unroll
                    for (int e = 0; e < 4; ++e) {
                        float f = vv[e];
                        __bf16 h = (__bf16)f;
                        Bhi[t][c][q * 4 + e] = h;
                        Blo[t][c][q * 4 + e] = (__bf16)(f - (float)h);
                    }
                }
            }
        }

        float best[2] = {-3.402823466e38f, -3.402823466e38f};
        int   bk[2]   = {0, 0};

        // ---- Loop over 16 codeword tiles; score = x.c - 0.5*||c||^2 ----
        for (int kt = 0; kt < 16; ++kt) {
            v8f acc[2];
#pragma unroll
            for (int v = 0; v < 8; ++v) {
                float c2v = -0.5f * c2s[kt * 16 + hl * 8 + v];
                acc[0][v] = c2v;
                acc[1][v] = c2v;
            }
            const int krow = kt * 16 + lm;
#pragma unroll
            for (int c = 0; c < 3; ++c) {
                // A layout (16-bit 16x32): lane holds row M=lane&15,
                // K = 16*hl + e (e = 0..15), matching B.
                const __bf16* h0p = &cb_hi[krow * SUBD + c * 32 + hl * 16];
                const __bf16* l0p = &cb_lo[krow * SUBD + c * 32 + hl * 16];
                v8bf h0 = *(const v8bf*)h0p;
                v8bf h1 = *(const v8bf*)(h0p + 8);
                v8bf l0 = *(const v8bf*)l0p;
                v8bf l1 = *(const v8bf*)(l0p + 8);
                v16bf Ahi, Alo;
#pragma unroll
                for (int e = 0; e < 8; ++e) {
                    Ahi[e] = h0[e]; Ahi[e + 8] = h1[e];
                    Alo[e] = l0[e]; Alo[e + 8] = l1[e];
                }
#pragma unroll
                for (int t = 0; t < 2; ++t) {
                    acc[t] = __builtin_amdgcn_wmma_f32_16x16x32_bf16(
                        false, Ahi, false, Bhi[t][c], (short)0, acc[t], false, false);
                    acc[t] = __builtin_amdgcn_wmma_f32_16x16x32_bf16(
                        false, Ahi, false, Blo[t][c], (short)0, acc[t], false, false);
                    acc[t] = __builtin_amdgcn_wmma_f32_16x16x32_bf16(
                        false, Alo, false, Bhi[t][c], (short)0, acc[t], false, false);
                }
            }
            // D layout: lane holds column n=lane&15; VGPR v is k = kt*16 + 8*hl + v.
            // Argmax over k is in-lane -> running best, ascending k keeps first-tie.
#pragma unroll
            for (int t = 0; t < 2; ++t) {
#pragma unroll
                for (int v = 0; v < 8; ++v) {
                    float s = acc[t][v];
                    if (s > best[t]) { best[t] = s; bk[t] = kt * 16 + hl * 8 + v; }
                }
            }
        }

        // ---- Combine the two lane halves, then gather winning codewords ----
#pragma unroll
        for (int t = 0; t < 2; ++t) {
            float ob  = __shfl_xor(best[t], 16, 32);
            int   obk = __shfl_xor(bk[t], 16, 32);
            if (ob > best[t] || (ob == best[t] && obk < bk[t])) {
                best[t] = ob; bk[t] = obk;
            }
            for (int r = 0; r < 16; ++r) {
                int kr = __shfl(bk[t], r, 32);
                const float* srcv = cb + ((size_t)m * NCODE + kr) * SUBD;
                float* dst = out + (size_t)(rowbase[t] + r) * EDIM + m * SUBD;
#pragma unroll
                for (int j = 0; j < 3; ++j)
                    dst[lane + j * 32] = srcv[lane + j * 32];
            }
        }
    }
}

extern "C" void kernel_launch(void* const* d_in, const int* in_sizes, int n_in,
                              void* d_out, int out_size, void* d_ws, size_t ws_size,
                              hipStream_t stream) {
    const float* emb = (const float*)d_in[0];   // (16, 4096, 768) f32
    const float* cb  = (const float*)d_in[1];   // (8, 256, 96) f32
    float* out = (float*)d_out;                 // (16, 4096, 768) f32
    dim3 grid(256, NSUB);
    pq_wmma_kernel<<<grid, 128, 0, stream>>>(emb, cb, out);
}
